// TreeBottomUpStep_57827439674228
// MI455X (gfx1250) — matrix operand order repaired and checked by reading, not compiled, not run on hardware
//
#include <hip/hip_runtime.h>
#include <hip/hip_bf16.h>

#define H 256
#define NEG_SLOPE 0.01f
#define BSTRIDE 264   // padded LDS row stride in halves: 132 dwords, %64==4 -> conflict-free fragment reads

typedef __attribute__((ext_vector_type(16))) _Float16 v16h;
typedef __attribute__((ext_vector_type(8)))  float    v8f;
typedef __attribute__((ext_vector_type(2)))  _Float16 h2;

// ---------------- zero init (sum + count contiguous) ----------------
__global__ void zero_kernel(float* __restrict__ p, size_t n) {
    size_t i = (size_t)blockIdx.x * blockDim.x + threadIdx.x;
    size_t stride = (size_t)gridDim.x * blockDim.x;
    for (; i < n; i += stride) p[i] = 0.0f;
}

// ---------------- fp32 -> f16 weight convert ----------------
__global__ void cvt_kernel(const float* __restrict__ w, _Float16* __restrict__ wh, int n) {
    int i = blockIdx.x * blockDim.x + threadIdx.x;
    if (i < n) wh[i] = (_Float16)w[i];
}

// ---------------- edge scatter: one wave32 per edge ----------------
__global__ __launch_bounds__(256) void scatter_kernel(
    const float* __restrict__ hfeat,
    const long long* __restrict__ eidx,
    float* __restrict__ sum, float* __restrict__ cnt,
    int E) {
    int wave = blockIdx.x * (blockDim.x >> 5) + (threadIdx.x >> 5);
    int lane = threadIdx.x & 31;
    if (wave >= E) return;
    long long p = eidx[wave];             // parent
    long long c = eidx[(size_t)E + wave]; // child
    const float4* src = (const float4*)(hfeat + (size_t)c * H);
    float* dst = sum + (size_t)p * H;
    float4 v0 = src[lane];       // elems 4*lane..4*lane+3   (coalesced 512B)
    float4 v1 = src[lane + 32];  // elems 128+4*lane..       (coalesced 512B)
    int e0 = 4 * lane;
    unsafeAtomicAdd(dst + e0 + 0,       v0.x);
    unsafeAtomicAdd(dst + e0 + 1,       v0.y);
    unsafeAtomicAdd(dst + e0 + 2,       v0.z);
    unsafeAtomicAdd(dst + e0 + 3,       v0.w);
    unsafeAtomicAdd(dst + 128 + e0 + 0, v1.x);
    unsafeAtomicAdd(dst + 128 + e0 + 1, v1.y);
    unsafeAtomicAdd(dst + 128 + e0 + 2, v1.z);
    unsafeAtomicAdd(dst + 128 + e0 + 3, v1.w);
    if (lane == 0) unsafeAtomicAdd(cnt + p, 1.0f);
}

// ---------------- mean = sum / max(count,1), convert to f16 ----------------
__global__ __launch_bounds__(256) void mean_kernel(
    const float* __restrict__ sum, const float* __restrict__ cnt,
    _Float16* __restrict__ meanH, int nrows) {
    int row = blockIdx.x;
    if (row >= nrows) return;
    int t = threadIdx.x;
    float inv = 1.0f / fmaxf(cnt[row], 1.0f);
    meanH[(size_t)row * H + t] = (_Float16)(sum[(size_t)row * H + t] * inv);
}

// ---------------- WMMA GEMM: LDS-staged B, double-buffered staging,
//                  one-deep pipelined fragment loads ------------------------
// D = A(f16, nrows x 256) x Bw(f16 weights, row-major [j,k])^T + bias
// FINAL==false: z = leaky(acc + bias) -> f16 Zout
// FINAL==true : out = hin + (acc + bias) * (cnt>0)
template <bool FINAL>
__global__ __launch_bounds__(256) void gemm_kernel(
    const _Float16* __restrict__ A,
    const _Float16* __restrict__ Bw,
    const float* __restrict__ bias,
    _Float16* __restrict__ Zout,
    const float* __restrict__ hin,
    const float* __restrict__ cnt,
    float* __restrict__ out,
    int nrows) {
    __shared__ __align__(16) _Float16 Blds[16 * BSTRIDE];

    int wv   = threadIdx.x >> 5;
    int lane = threadIdx.x & 31;
    int tileM = blockIdx.x * (blockDim.x >> 5) + wv;
    bool valid = (tileM * 16 < nrows);
    int mbase  = valid ? tileM * 16 : 0;          // clamp; stay alive for barriers
    bool full  = valid && (mbase + 16 <= nrows);  // N % 16 == 0 -> always true here

    int m16  = lane & 15;
    int half = lane >> 4;
    int rowA = mbase + m16;
    if (rowA >= nrows) rowA = nrows - 1;          // safe read; stores predicated

    // Load A tile (16x256 f16) into 8 K-chunk fragments (coalesces to b128 pairs)
    v16h a[8];
#pragma unroll
    for (int kc = 0; kc < 8; ++kc) {
#pragma unroll
        for (int v = 0; v < 8; ++v) {
            int kk = kc * 32 + ((v < 4) ? (half * 8 + v * 2)
                                        : (16 + half * 8 + (v - 4) * 2));
            h2 pr = *(const h2*)(A + (size_t)rowA * H + kk);
            a[kc][2 * v]     = pr.x;
            a[kc][2 * v + 1] = pr.y;
        }
    }

    float mask[8];
    if (FINAL) {
        // Warm the residual tile: 16 rows x 1KB, 512B/lane -> global_prefetch_b8
        const float* hrow = hin + (size_t)rowA * H + half * 128;
        __builtin_prefetch(hrow +  0, 0, 1);
        __builtin_prefetch(hrow + 32, 0, 1);
        __builtin_prefetch(hrow + 64, 0, 1);
        __builtin_prefetch(hrow + 96, 0, 1);
#pragma unroll
        for (int r = 0; r < 8; ++r) {
            int m = mbase + half * 8 + r;
            if (m >= nrows) m = nrows - 1;
            mask[r] = (cnt[m] > 0.0f) ? 1.0f : 0.0f;
        }
    }

    // B-fragment loader (ISA 16-bit B layout; conflict-free banks via BSTRIDE pad)
    const _Float16* brow = &Blds[m16 * BSTRIDE + half * 8];
    auto load_frag = [&](int kc) -> v16h {
        v16h b;
#pragma unroll
        for (int v = 0; v < 8; ++v) {
            h2 pr = *(const h2*)(brow + kc * 32 + ((v < 4) ? v * 2 : 16 + (v - 4) * 2));
            b[2 * v]     = pr.x;
            b[2 * v + 1] = pr.y;
        }
        return b;
    };

    // Per-thread staging coords: 256 threads cover 16 rows x 256 halves (8KB tile)
    int srow = threadIdx.x >> 4;          // 0..15 : row within B tile
    int scol = (threadIdx.x & 15) * 16;   // halves: 16 per thread (32B, coalesced)
    const _Float16* gbase = Bw + (size_t)srow * H + scol;
    uint4* sp0 = (uint4*)&Blds[srow * BSTRIDE + scol];
    uint4* sp1 = (uint4*)&Blds[srow * BSTRIDE + scol + 8];

    // Prologue: tile 0 in registers
    uint4 g0 = *(const uint4*)(gbase);
    uint4 g1 = *(const uint4*)(gbase + 8);

    for (int tn = 0; tn < 16; ++tn) {
        __syncthreads();                  // previous tile fully consumed
        *sp0 = g0;                        // ds_store_b128 x2
        *sp1 = g1;
        if (tn < 15) {                    // issue next tile's global loads early;
            const _Float16* gn = gbase + (size_t)(tn + 1) * 16 * H;
            g0 = *(const uint4*)(gn);     // latency hidden under the 8 WMMAs below
            g1 = *(const uint4*)(gn + 8);
        }
        __syncthreads();                  // staged tile visible

        // One-deep pipeline: WMMA kc consumes fragment loaded one step earlier,
        // so each wmma waits on the older ds_load pair (partial s_wait_dscnt).
        v16h bcur = load_frag(0);
        v8f acc = {};
#pragma unroll
        for (int kc = 0; kc < 8; ++kc) {
            v16h bnext = (kc < 7) ? load_frag(kc + 1) : bcur;
            acc = __builtin_amdgcn_wmma_f32_16x16x32_f16(
                false, a[kc], false, bcur, (short)0, acc, false, false);
            bcur = bnext;
        }

        int nB = tn * 16 + m16;           // output column for this lane
        float bv = bias[nB];
        if (full) {
#pragma unroll
            for (int r = 0; r < 8; ++r) {
                int m = mbase + half * 8 + r;   // C/D layout: VGPR r -> row half*8+r
                float z = acc[r] + bv;
                if (!FINAL) {
                    z = (z >= 0.0f) ? z : NEG_SLOPE * z;
                    Zout[(size_t)m * H + nB] = (_Float16)z;
                } else {
                    out[(size_t)m * H + nB] =
                        hin[(size_t)m * H + nB] + z * mask[r];
                }
            }
        } else if (valid) {               // cold generic tail (never taken: N%16==0)
#pragma unroll
            for (int r = 0; r < 8; ++r) {
                int m = mbase + half * 8 + r;
                if (m >= nrows) continue;
                float z = acc[r] + bv;
                if (!FINAL) {
                    z = (z >= 0.0f) ? z : NEG_SLOPE * z;
                    Zout[(size_t)m * H + nB] = (_Float16)z;
                } else {
                    out[(size_t)m * H + nB] =
                        hin[(size_t)m * H + nB] + z * mask[r];
                }
            }
        }
    }
}

extern "C" void kernel_launch(void* const* d_in, const int* in_sizes, int n_in,
                              void* d_out, int out_size, void* d_ws, size_t ws_size,
                              hipStream_t stream) {
    (void)n_in; (void)out_size; (void)ws_size;
    const float*     hfeat = (const float*)d_in[0];
    const long long* eidx  = (const long long*)d_in[1];
    const float*     W1    = (const float*)d_in[2];
    const float*     b1    = (const float*)d_in[3];
    const float*     W2    = (const float*)d_in[4];
    const float*     b2    = (const float*)d_in[5];
    float*           outp  = (float*)d_out;

    const int N = in_sizes[0] / H;
    const int E = in_sizes[1] / 2;

    char* ws = (char*)d_ws;
    float*     sum   = (float*)ws;                              // N*H f32
    float*     cnt   = (float*)(ws + (size_t)N * H * 4);        // N f32
    _Float16*  meanH = (_Float16*)(ws + (size_t)N * H * 4 + (size_t)N * 4);
    _Float16*  zH    = meanH + (size_t)N * H;
    _Float16*  w1h   = zH + (size_t)N * H;                      // H*H f16
    _Float16*  w2h   = w1h + H * H;

    // 1) zero sum+count (contiguous)
    zero_kernel<<<2048, 256, 0, stream>>>(sum, (size_t)N * H + (size_t)N);
    // 2) weights to f16 (L2-resident B operands)
    cvt_kernel<<<(H * H + 255) / 256, 256, 0, stream>>>(W1, w1h, H * H);
    cvt_kernel<<<(H * H + 255) / 256, 256, 0, stream>>>(W2, w2h, H * H);
    // 3) edge scatter: one wave per edge, 8 waves per block
    scatter_kernel<<<(E + 7) / 8, 256, 0, stream>>>(hfeat, eidx, sum, cnt, E);
    // 4) mean -> f16
    mean_kernel<<<N, 256, 0, stream>>>(sum, cnt, meanH, N);
    // 5) MLP layer 1 (WMMA) + LeakyReLU -> f16
    int mtiles  = (N + 15) / 16;
    int gblocks = (mtiles + 7) / 8;
    gemm_kernel<false><<<gblocks, 256, 0, stream>>>(meanH, w1h, b1, zH,
                                                    nullptr, nullptr, nullptr, N);
    // 6) MLP layer 2 (WMMA) + masked residual -> f32 out
    gemm_kernel<true><<<gblocks, 256, 0, stream>>>(zH, w2h, b2, nullptr,
                                                   hfeat, cnt, outp, N);
}
